// Local_Layer_72103910965517
// MI455X (gfx1250) — matrix operand order repaired and checked by reading, not compiled, not run on hardware
//
#include <hip/hip_runtime.h>

typedef __attribute__((ext_vector_type(16))) _Float16 v16h;
typedef __attribute__((ext_vector_type(8)))  float    v8f;
typedef __attribute__((ext_vector_type(4)))  int      v4i;

typedef __attribute__((address_space(1))) v4i* g4p;   // global v4i*
typedef __attribute__((address_space(3))) v4i* l4p;   // LDS v4i*

#define GAT_M   32
#define GAT_H   8
#define GAT_D   16
#define GAT_HID 128
#define MPADF   132     // padded row stride (floats): 528B, 16B aligned, bank-staggered
#define WPB     3       // waves (=nodes) per block
#define BLOCK   (WPB*32)

struct __align__(16) WaveLds {
    float kv[GAT_M][MPADF];       // row-major f32 K=V tile (async-staged)
    float attn[GAT_H][GAT_M];     // softmax weights, lane==m layout
    float outbuf[GAT_HID];        // staging for coalesced store
    float qd[GAT_H];              // q . W1 per head
};

__global__ __launch_bounds__(BLOCK)
void gat_attn_kernel(const float* __restrict__ h, const float* __restrict__ s,
                     const float* __restrict__ W, const float* __restrict__ b,
                     float* __restrict__ out, int n)
{
    __shared__ WaveLds wl[WPB];
    __shared__ float w_lds[GAT_H][2*GAT_D];
    __shared__ float b_lds[GAT_H];

    const int tid = threadIdx.x;
    for (int i = tid; i < GAT_H*2*GAT_D; i += BLOCK) ((float*)w_lds)[i] = W[i];
    if (tid < GAT_H) b_lds[tid] = b[tid];
    __syncthreads();

    const int wave = tid >> 5;
    const int lane = tid & 31;
    const int node = blockIdx.x * WPB + wave;
    if (node >= n) return;                       // wave-uniform exit (after barrier)

    WaveLds& L = wl[wave];
    const float* sp = s + (size_t)node * (GAT_M * GAT_HID);

    // ---- Phase 1: stream 16KB K=V tile straight into LDS.
    // gfx1250 async global->LDS DMA (ASYNCcnt), one b128 per lane per row.
#if __has_builtin(__builtin_amdgcn_global_load_async_to_lds_b128)
    #pragma unroll 4
    for (int m = 0; m < GAT_M; ++m) {
        __builtin_amdgcn_global_load_async_to_lds_b128(
            (g4p)(void*)(sp + m*GAT_HID + lane*4),
            (l4p)(void*)(&L.kv[m][lane*4]),
            /*offset=*/0, /*cpol=*/0);
    }
#else
    #pragma unroll 4
    for (int m = 0; m < GAT_M; ++m) {
        const float4 v = *(const float4*)(sp + m*GAT_HID + lane*4);
        *(float4*)(&L.kv[m][lane*4]) = v;
    }
#endif

    // ---- Phase 1b: qd[head] = sum_d q[head,d] * W[head,d]   (W1 half, coalesced)
    {
        const float4 hv = *(const float4*)(h + (size_t)node*GAT_HID + lane*4);
        const int head = lane >> 2;
        const int d0   = (lane & 3) * 4;
        float p = hv.x*w_lds[head][d0+0] + hv.y*w_lds[head][d0+1]
                + hv.z*w_lds[head][d0+2] + hv.w*w_lds[head][d0+3];
        p += __shfl_xor(p, 1);
        p += __shfl_xor(p, 2);
        if ((lane & 3) == 0) L.qd[head] = p;
    }

    // Drain the async tile DMA before consuming it.
#if __has_builtin(__builtin_amdgcn_s_wait_asynccnt)
    __builtin_amdgcn_s_wait_asynccnt(0);
#else
    asm volatile("s_wait_asynccnt 0x0" ::: "memory");
#endif
    __builtin_amdgcn_wave_barrier();

    // ---- Phase 2: scores + softmax over M, per head; lane == m (f32 math)
    for (int i = 0; i < GAT_H; ++i) {
        const float4* krow = (const float4*)(&L.kv[lane][i*GAT_D]);
        float sc = L.qd[i] + b_lds[i];
        #pragma unroll
        for (int q4 = 0; q4 < 4; ++q4) {
            const float4 kq = krow[q4];
            sc += kq.x * w_lds[i][GAT_D + q4*4 + 0]
                + kq.y * w_lds[i][GAT_D + q4*4 + 1]
                + kq.z * w_lds[i][GAT_D + q4*4 + 2]
                + kq.w * w_lds[i][GAT_D + q4*4 + 3];
        }
        sc = fmaxf(sc, 0.0f);                    // ReLU
        float mx = sc;
        #pragma unroll
        for (int off = 16; off > 0; off >>= 1)
            mx = fmaxf(mx, __shfl_xor(mx, off));
        const float e = __expf(sc - mx);
        float sum = e;
        #pragma unroll
        for (int off = 16; off > 0; off >>= 1)
            sum += __shfl_xor(sum, off);
        L.attn[i][lane] = e / sum;
    }
    __builtin_amdgcn_wave_barrier();

    // ---- Phase 3: out_h[d] = sum_m attn[m,h] * K[m,h,d] as V_WMMA_F32_16X16X32_F16
    // A[16x32]: every row = attn_h; all C rows equal -> C[0] valid on ALL lanes
    //   (lanes 0-15: M=0,N=lane; lanes 16-31: M=8,N=lane-16), so no exec masking.
    // f16 A layout (matches B): lanes0-15 hold K=0..15, lanes16-31 K=16..31.
    // f16 B 32x16 (this code's contiguous order, not ISA table):  lanes0-15 hold K=0..15 (2 rows/VGPR), lanes16-31 K=16..31.
    const int half  = lane >> 4;
    const int ncol  = lane & 15;
    const int kbase = half ? 16 : 0;
    for (int i = 0; i < GAT_H; ++i) {
        v16h A, B;
        #pragma unroll
        for (int j = 0; j < 8; ++j) {
            const int kk = kbase + 2*j;
            A[2*j+0] = (_Float16)L.attn[i][kk+0];
            A[2*j+1] = (_Float16)L.attn[i][kk+1];
        }
        const int col = i*GAT_D + ncol;
        #pragma unroll
        for (int j = 0; j < 8; ++j) {
            B[2*j+0] = (_Float16)L.kv[kbase + 2*j + 0][col];
            B[2*j+1] = (_Float16)L.kv[kbase + 2*j + 1][col];
        }
        v8f C = {};
        C = __builtin_amdgcn_wmma_f32_16x16x32_f16(
                /*neg_a=*/false, A, /*neg_b=*/false, B,
                /*c_mod=*/(short)0, C, /*reuse_a=*/false, /*reuse_b=*/false);
        L.outbuf[i*GAT_D + ncol] = C[0];         // both half-waves write same value
    }
    __builtin_amdgcn_wave_barrier();

    // ---- Phase 4: coalesced b128 store
    const float4 o = *(const float4*)(&L.outbuf[lane*4]);
    *(float4*)(out + (size_t)node*GAT_HID + lane*4) = o;
}

extern "C" void kernel_launch(void* const* d_in, const int* in_sizes, int n_in,
                              void* d_out, int out_size, void* d_ws, size_t ws_size,
                              hipStream_t stream) {
    const float* h = (const float*)d_in[0];
    const float* s = (const float*)d_in[1];
    const float* W = (const float*)d_in[2];
    const float* b = (const float*)d_in[3];
    float* out     = (float*)d_out;
    const int n    = in_sizes[0] / GAT_HID;      // N nodes
    const int blocks = (n + WPB - 1) / WPB;
    gat_attn_kernel<<<blocks, BLOCK, 0, stream>>>(h, s, W, b, out, n);
}
